// AttentionPooling_10771777978565
// MI455X (gfx1250) — matrix operand order repaired and checked by reading, not compiled, not run on hardware
//
#include <hip/hip_runtime.h>
#include <hip/hip_fp16.h>
#include <math.h>

typedef __attribute__((ext_vector_type(16))) _Float16 v16h;
typedef __attribute__((ext_vector_type(8)))  _Float16 v8h;
typedef __attribute__((ext_vector_type(4)))  _Float16 v4h;
typedef __attribute__((ext_vector_type(8)))  float    v8f;
typedef __attribute__((ext_vector_type(4)))  float    v4f;

#define DEVI __device__ __forceinline__

// ---------------- WMMA helpers (gfx1250 wave32) ----------------

DEVI v8f wmma_f16(v16h a, v16h b, v8f c) {
  // v_wmma_f32_16x16x32_f16 : D = A(16x32) * B(32x16) + C(16x16 f32)
  return __builtin_amdgcn_wmma_f32_16x16x32_f16(false, a, false, b, (short)0, c, false, false);
}

// A-fragment (16x32 f16) from row-major f16 LDS, row stride `stride` elems.
// lane<16 holds row lane,    K = {k0..k0+15}
// lane>=16 holds row lane-16, K = {k0+16..k0+31}
DEVI v16h ldsA(const _Float16* base, int stride, int m0, int k0, int lane) {
  int row = m0 + (lane & 15);
  int kb  = k0 + ((lane >> 4) << 4);
  const _Float16* p = base + row * stride + kb;
  v8h lo = *(const v8h*)p;          // K offsets +0..7
  v8h hi = *(const v8h*)(p + 8);    // K offsets +8..15
  v16h a;
#pragma unroll
  for (int i = 0; i < 8; ++i) { a[i] = lo[i]; a[i + 8] = hi[i]; }
  return a;
}

// One 16x16 output tile: acc = bias; acc += sum_kt A[kt] * Bblob[nt][kt]
template <int NKT>
DEVI v8f mmtile(const v16h* aF, const v16h* bBlob, int nt, int lane, float bias) {
  v8f acc;
#pragma unroll
  for (int r = 0; r < 8; ++r) acc[r] = bias;
#pragma unroll
  for (int kt = 0; kt < NKT; ++kt) {
    v16h b = bBlob[((nt * NKT + kt) << 5) + lane];   // 32B/lane, 1KB/wave coalesced
    acc = wmma_f16(aF[kt], b, acc);
  }
  return acc;
}

// C/D layout: lane holds column n = lane&15; VGPR r holds row m0 + (lane>=16?8:0) + r
DEVI void stTileH(_Float16* dst, int stride, int m0, int n0, int lane, v8f acc) {
  int n  = n0 + (lane & 15);
  int mb = m0 + ((lane >> 4) << 3);
#pragma unroll
  for (int r = 0; r < 8; ++r) dst[(mb + r) * stride + n] = (_Float16)acc[r];
}

DEVI void stTileAddH(_Float16* dst, int stride, int m0, int n0, int lane, v8f acc) {
  int n  = n0 + (lane & 15);
  int mb = m0 + ((lane >> 4) << 3);
#pragma unroll
  for (int r = 0; r < 8; ++r) {
    int i = (mb + r) * stride + n;
    dst[i] = (_Float16)((float)dst[i] + acc[r]);
  }
}

DEVI float wred32(float v) {
#pragma unroll
  for (int m = 16; m >= 1; m >>= 1) v += __shfl_xor(v, m, 32);
  return v;
}

// Vectorized LayerNorm over a 384-elem f16 LDS row, in place (one wave per row).
// Lane handles 6 __half2 pairs at pair-index lane + j*32.
DEVI void ln384v(_Float16* row, const float* __restrict__ g, const float* __restrict__ b, int lane) {
  __half2* rp = (__half2*)row;
  const float2* g2 = (const float2*)g;
  const float2* b2 = (const float2*)b;
  float2 v[6];
  float s = 0.f, s2 = 0.f;
#pragma unroll
  for (int j = 0; j < 6; ++j) {
    int pi = lane + j * 32;
    v[j] = __half22float2(rp[pi]);
    s  += v[j].x + v[j].y;
    s2 += v[j].x * v[j].x + v[j].y * v[j].y;
  }
  s = wred32(s); s2 = wred32(s2);
  float mean = s * (1.f / 384.f);
  float rstd = rsqrtf(s2 * (1.f / 384.f) - mean * mean + 1e-5f);
#pragma unroll
  for (int j = 0; j < 6; ++j) {
    int pi = lane + j * 32;
    float2 gg = g2[pi], bb = b2[pi];
    rp[pi] = __floats2half2_rn((v[j].x - mean) * rstd * gg.x + bb.x,
                               (v[j].y - mean) * rstd * gg.y + bb.y);
  }
}

// Prefetch `bytes` starting at p, one 128B line per lane round-robin
DEVI void pfetch(const void* p, int bytes, int lane) {
  const char* c = (const char*)p;
  for (int o = lane * 128; o < bytes; o += 32 * 128) __builtin_prefetch(c + o, 0, 1);
}

// ---------------- weight pre-pack: f32 (N,K) row-major -> f16 B-fragment blob ----------------
// blob element t = ((nt*nkt+kt)*32 + lane)*16 + e  <->  W[nt*16 + lane%16][kt*32 + (lane/16)*16 + e]
__global__ void pack_w_kernel(const float* __restrict__ src, _Float16* __restrict__ dst,
                              int N, int K, int rowOff) {
  int t = blockIdx.x * 256 + threadIdx.x;
  if (t >= N * K) return;
  int nkt  = K >> 5;
  int e    = t & 15;
  int lane = (t >> 4) & 31;
  int ft   = t >> 9;
  int kt   = ft % nkt;
  int nt   = ft / nkt;
  int n    = nt * 16 + (lane & 15);
  int k    = kt * 32 + ((lane >> 4) << 4) + e;
  dst[t] = (_Float16)src[(size_t)(rowOff + n) * K + k];
}

// ---------------- fused attention-pooling block: 16 batch items per workgroup ----------------
// Packed blob offsets (f16 elems)
#define OFF_WV   0u
#define OFF_WOC  147456u
#define OFF_WQKV 294912u
#define OFF_WOS  737280u
#define OFF_W1   884736u
#define OFF_W2   1474560u
#define OFF_WG   2064384u

__launch_bounds__(256, 1)
__global__ void fused_attn_pool(
    const float* __restrict__ x, const float* __restrict__ latents,
    const float* __restrict__ ca_b_in, const float* __restrict__ ca_b_out,
    const float* __restrict__ sa_b_in, const float* __restrict__ sa_b_out,
    const float* __restrict__ n1g, const float* __restrict__ n1b,
    const float* __restrict__ n2g, const float* __restrict__ n2b,
    const float* __restrict__ n3g, const float* __restrict__ n3b,
    const float* __restrict__ fb1, const float* __restrict__ fb2,
    const float* __restrict__ gb,
    const _Float16* __restrict__ wpack, float* __restrict__ out)
{
  extern __shared__ char smem[];
  _Float16* BIG = (_Float16*)smem;                    // 192KB region, multi-use
  _Float16* H1  = (_Float16*)(smem + 192 * 1024);     // 48KB: 64 rows x 384

  const int tid = threadIdx.x, lane = tid & 31, w = tid >> 5;
  const int b0 = blockIdx.x * 16;

  const v16h* WV   = (const v16h*)(wpack + OFF_WV);
  const v16h* WOC  = (const v16h*)(wpack + OFF_WOC);
  const v16h* WQKV = (const v16h*)(wpack + OFF_WQKV);
  const v16h* WOS  = (const v16h*)(wpack + OFF_WOS);
  const v16h* W1   = (const v16h*)(wpack + OFF_W1);
  const v16h* W2   = (const v16h*)(wpack + OFF_W2);
  const v16h* WG   = (const v16h*)(wpack + OFF_WG);

  _Float16* X  = BIG;            // 16x384 f16
  _Float16* T  = BIG + 6144;     // 16x384 f16
  _Float16* CA = BIG + 12288;    // 16x384 f16

  // ---- stage x tile (16x384 f32) -> f16 LDS (non-temporal: stream-once data)
  {
    const v4f* xs = (const v4f*)(x + (size_t)b0 * 384);
#pragma unroll
    for (int i = 0; i < 6; ++i) {
      int idx = tid + i * 256;                  // 1536 v4f total
      v4f v = __builtin_nontemporal_load(xs + idx);
      v4h h = {(_Float16)v.x, (_Float16)v.y, (_Float16)v.z, (_Float16)v.w};
      *(v4h*)(X + idx * 4) = h;
    }
  }
  __syncthreads();

  // ---- cross-attention collapses (softmax over 1 key == 1): ca = (x@Wv^T+bv)@Wo^T+bo
  {
    v16h aF[12];
#pragma unroll
    for (int kt = 0; kt < 12; ++kt) aF[kt] = ldsA(X, 384, 0, kt * 32, lane);
    for (int i = 0; i < 3; ++i) {
      int nt = w * 3 + i;
      v8f acc = mmtile<12>(aF, WV, nt, lane, ca_b_in[768 + nt * 16 + (lane & 15)]);
      stTileH(T, 384, 0, nt * 16, lane, acc);
    }
  }
  __syncthreads();
  {
    v16h aF[12];
#pragma unroll
    for (int kt = 0; kt < 12; ++kt) aF[kt] = ldsA(T, 384, 0, kt * 32, lane);
    for (int i = 0; i < 3; ++i) {
      int nt = w * 3 + i;
      v8f acc = mmtile<12>(aF, WOC, nt, lane, ca_b_out[nt * 16 + (lane & 15)]);
      stTileH(CA, 384, 0, nt * 16, lane, acc);
    }
  }
  __syncthreads();

  // ---- LN1: h1[bi*4+l] = LN(ca[bi] + latents[l])   (8 rows per wave, vectorized)
  for (int rr = 0; rr < 8; ++rr) {
    int ro = w * 8 + rr;                 // 0..63
    int bi = ro >> 2, l = ro & 3;
    const __half2* sp = (const __half2*)(CA + bi * 384);
    const float2*  lp = (const float2*)(latents + l * 384);
    __half2*       dp = (__half2*)(H1 + ro * 384);
    const float2* g2 = (const float2*)n1g;
    const float2* b2 = (const float2*)n1b;
    float2 v[6];
    float s = 0.f, s2 = 0.f;
#pragma unroll
    for (int j = 0; j < 6; ++j) {
      int pi = lane + j * 32;
      float2 c = __half22float2(sp[pi]);
      float2 la = lp[pi];
      v[j].x = c.x + la.x; v[j].y = c.y + la.y;
      s  += v[j].x + v[j].y;
      s2 += v[j].x * v[j].x + v[j].y * v[j].y;
    }
    s = wred32(s); s2 = wred32(s2);
    float mean = s * (1.f / 384.f);
    float rstd = rsqrtf(s2 * (1.f / 384.f) - mean * mean + 1e-5f);
#pragma unroll
    for (int j = 0; j < 6; ++j) {
      int pi = lane + j * 32;
      float2 gg = g2[pi], bb = b2[pi];
      dp[pi] = __floats2half2_rn((v[j].x - mean) * rstd * gg.x + bb.x,
                                 (v[j].y - mean) * rstd * gg.y + bb.y);
    }
  }
  __syncthreads();

  _Float16* QKV = BIG;   // 64 x 1152 f16 (q | k | v)

  // ---- self-attention qkv projection (64x384 @ 384x1152)
  {
    int mt = w & 3, ntBase = (w >> 2) * 36;
    v16h aF[12];
#pragma unroll
    for (int kt = 0; kt < 12; ++kt) aF[kt] = ldsA(H1, 384, mt * 16, kt * 32, lane);
    for (int i = 0; i < 36; ++i) {
      int nt = ntBase + i;
      v8f acc = mmtile<12>(aF, WQKV, nt, lane, sa_b_in[nt * 16 + (lane & 15)]);
      stTileH(QKV, 1152, mt * 16, nt * 16, lane, acc);
    }
  }
  __syncthreads();

  // ---- attention core: L=4, H=8, dh=48 ; lane = head*4 + qrow ; o overwrites q slots
  // vectorized: rows moved as v8h (ds_load_b128)
  {
    const float scale = 0.14433756729740645f;   // 1/sqrt(48)
    for (int ii = 0; ii < 2; ++ii) {
      int bi = w * 2 + ii;
      int head = lane >> 2, qi = lane & 3;
      const _Float16* qp = QKV + (bi * 4 + qi) * 1152 + head * 48;
      v8h qv[6];
#pragma unroll
      for (int c = 0; c < 6; ++c) qv[c] = *(const v8h*)(qp + c * 8);
      float s[4];
#pragma unroll
      for (int j = 0; j < 4; ++j) {
        const _Float16* kp = QKV + (bi * 4 + j) * 1152 + 384 + head * 48;
        float acc = 0.f;
#pragma unroll
        for (int c = 0; c < 6; ++c) {
          v8h kv = *(const v8h*)(kp + c * 8);
#pragma unroll
          for (int e = 0; e < 8; ++e) acc += (float)qv[c][e] * (float)kv[e];
        }
        s[j] = acc * scale;
      }
      float mx = fmaxf(fmaxf(s[0], s[1]), fmaxf(s[2], s[3]));
      float e0 = expf(s[0] - mx), e1 = expf(s[1] - mx);
      float e2 = expf(s[2] - mx), e3 = expf(s[3] - mx);
      float inv = 1.f / (e0 + e1 + e2 + e3);
      float a0 = e0 * inv, a1 = e1 * inv, a2 = e2 * inv, a3 = e3 * inv;
      const _Float16* v0 = QKV + (bi * 4 + 0) * 1152 + 768 + head * 48;
      const _Float16 *v1 = v0 + 1152, *v2 = v0 + 2304, *v3 = v0 + 3456;
      _Float16* op = QKV + (bi * 4 + qi) * 1152 + head * 48;   // q no longer needed
#pragma unroll
      for (int c = 0; c < 6; ++c) {
        v8h a = *(const v8h*)(v0 + c * 8);
        v8h b = *(const v8h*)(v1 + c * 8);
        v8h cc = *(const v8h*)(v2 + c * 8);
        v8h d = *(const v8h*)(v3 + c * 8);
        v8h o;
#pragma unroll
        for (int e = 0; e < 8; ++e)
          o[e] = (_Float16)(a0 * (float)a[e] + a1 * (float)b[e] +
                            a2 * (float)cc[e] + a3 * (float)d[e]);
        *(v8h*)(op + c * 8) = o;
      }
    }
    // prefetch out-projection weights for this wave's tile range
    pfetch(WOS + (size_t)((w >> 2) * 12) * 12 * 32, 12 * 1024, lane);
  }
  __syncthreads();

  // ---- out projection + residual into H1
  {
    int mt = w & 3, ntBase = (w >> 2) * 12;
    v16h aF[12];
#pragma unroll
    for (int kt = 0; kt < 12; ++kt) aF[kt] = ldsA(QKV, 1152, mt * 16, kt * 32, lane);
    for (int i = 0; i < 12; ++i) {
      int nt = ntBase + i;
      v8f acc = mmtile<12>(aF, WOS, nt, lane, sa_b_out[nt * 16 + (lane & 15)]);
      stTileAddH(H1, 384, mt * 16, nt * 16, lane, acc);
    }
  }
  __syncthreads();

  // ---- LN2 in place; prefetch FFN1 weights
  for (int rr = 0; rr < 8; ++rr) ln384v(H1 + (w * 8 + rr) * 384, n2g, n2b, lane);
  pfetch(W1 + (size_t)((w >> 2) * 48) * 12 * 32, 12 * 1024, lane);
  __syncthreads();

  _Float16* FF = BIG;   // 64 x 1536 f16

  // ---- FFN1 + exact GELU
  {
    int mt = w & 3, ntBase = (w >> 2) * 48;
    v16h aF[12];
#pragma unroll
    for (int kt = 0; kt < 12; ++kt) aF[kt] = ldsA(H1, 384, mt * 16, kt * 32, lane);
    for (int i = 0; i < 48; ++i) {
      int nt = ntBase + i;
      v8f acc = mmtile<12>(aF, W1, nt, lane, fb1[nt * 16 + (lane & 15)]);
#pragma unroll
      for (int r = 0; r < 8; ++r) {
        float xg = acc[r];
        acc[r] = 0.5f * xg * (1.0f + erff(xg * 0.70710678118f));
      }
      stTileH(FF, 1536, mt * 16, nt * 16, lane, acc);
    }
  }
  __syncthreads();

  // ---- FFN2 (K=1536 = 48 k-tiles), register-blocked: 6 output tiles live, A-frags reused
  {
    int mt = w & 3, ntBase = (w >> 2) * 12;
    for (int c = 0; c < 2; ++c) {
      v8f acc[6];
#pragma unroll
      for (int q = 0; q < 6; ++q) {
        float bias = fb2[(ntBase + c * 6 + q) * 16 + (lane & 15)];
#pragma unroll
        for (int r = 0; r < 8; ++r) acc[q][r] = bias;
      }
      for (int kb = 0; kb < 4; ++kb) {
        v16h aF[12];
#pragma unroll
        for (int kt = 0; kt < 12; ++kt) aF[kt] = ldsA(FF, 1536, mt * 16, (kb * 12 + kt) * 32, lane);
#pragma unroll
        for (int q = 0; q < 6; ++q) {
          int nt = ntBase + c * 6 + q;
#pragma unroll
          for (int kt = 0; kt < 12; ++kt) {
            v16h bfr = W2[((nt * 48 + kb * 12 + kt) << 5) + lane];
            acc[q] = wmma_f16(aF[kt], bfr, acc[q]);
          }
        }
      }
#pragma unroll
      for (int q = 0; q < 6; ++q)
        stTileAddH(H1, 384, mt * 16, (ntBase + c * 6 + q) * 16, lane, acc[q]);
    }
  }
  __syncthreads();

  // ---- LN3 in place; prefetch gate weights
  for (int rr = 0; rr < 8; ++rr) ln384v(H1 + (w * 8 + rr) * 384, n3g, n3b, lane);
  pfetch(WG + (size_t)(w * 3) * 12 * 32, 12 * 1024, lane);
  __syncthreads();

  // ---- pool over L=4 (f32 + f16 copies; FF region now free), vectorized
  float*    PF = (float*)smem;                       // 16x384 f32
  _Float16* PH = (_Float16*)(smem + 24 * 1024);      // 16x384 f16
#pragma unroll
  for (int i = 0; i < 12; ++i) {
    int p = tid + i * 256;                           // pair index 0..3071
    int bi = p / 192, dp2 = p - bi * 192;
    const __half2* r = (const __half2*)H1 + (size_t)bi * 4 * 192 + dp2;
    float2 a0 = __half22float2(r[0]);
    float2 a1 = __half22float2(r[192]);
    float2 a2 = __half22float2(r[384]);
    float2 a3 = __half22float2(r[576]);
    float px = (a0.x + a1.x + a2.x + a3.x) * 0.25f;
    float py = (a0.y + a1.y + a2.y + a3.y) * 0.25f;
    ((float2*)PF)[p] = make_float2(px, py);
    ((__half2*)PH)[p] = __floats2half2_rn(px, py);
  }
  __syncthreads();

  // ---- gate GEMM + sigmoid + final output (non-temporal stores)
  {
    v16h aF[12];
#pragma unroll
    for (int kt = 0; kt < 12; ++kt) aF[kt] = ldsA(PH, 384, 0, kt * 32, lane);
    for (int i = 0; i < 3; ++i) {
      int nt = w * 3 + i;
      v8f acc = mmtile<12>(aF, WG, nt, lane, gb[nt * 16 + (lane & 15)]);
      int n  = nt * 16 + (lane & 15);
      int mb = (lane >> 4) << 3;
#pragma unroll
      for (int r = 0; r < 8; ++r) {
        int m = mb + r;
        float p = PF[m * 384 + n];
        float g = 1.f / (1.f + expf(-acc[r]));
        __builtin_nontemporal_store(p * g, &out[(size_t)(b0 + m) * 384 + n]);
      }
    }
  }
}

// ---------------- host launch ----------------
extern "C" void kernel_launch(void* const* d_in, const int* in_sizes, int n_in,
                              void* d_out, int out_size, void* d_ws, size_t ws_size,
                              hipStream_t stream) {
  const float* x        = (const float*)d_in[0];
  const float* latents  = (const float*)d_in[1];
  const float* ca_w_in  = (const float*)d_in[2];
  const float* ca_b_in  = (const float*)d_in[3];
  const float* ca_w_out = (const float*)d_in[4];
  const float* ca_b_out = (const float*)d_in[5];
  const float* sa_w_in  = (const float*)d_in[6];
  const float* sa_b_in  = (const float*)d_in[7];
  const float* sa_w_out = (const float*)d_in[8];
  const float* sa_b_out = (const float*)d_in[9];
  const float* n1g = (const float*)d_in[10];
  const float* n1b = (const float*)d_in[11];
  const float* n2g = (const float*)d_in[12];
  const float* n2b = (const float*)d_in[13];
  const float* n3g = (const float*)d_in[14];
  const float* n3b = (const float*)d_in[15];
  const float* fw1 = (const float*)d_in[16];
  const float* fb1 = (const float*)d_in[17];
  const float* fw2 = (const float*)d_in[18];
  const float* fb2 = (const float*)d_in[19];
  const float* gw  = (const float*)d_in[20];
  const float* gb  = (const float*)d_in[21];

  _Float16* wp = (_Float16*)d_ws;   // 2,211,840 f16 = ~4.4 MB of workspace

  auto pack = [&](const float* src, size_t dstOff, int N, int K, int rowOff) {
    int total = N * K;
    pack_w_kernel<<<(total + 255) / 256, 256, 0, stream>>>(src, wp + dstOff, N, K, rowOff);
  };
  pack(ca_w_in,  OFF_WV,   384,  384, 768);   // Wv slice of ca_w_in (rows 768..1151)
  pack(ca_w_out, OFF_WOC,  384,  384, 0);
  pack(sa_w_in,  OFF_WQKV, 1152, 384, 0);
  pack(sa_w_out, OFF_WOS,  384,  384, 0);
  pack(fw1,      OFF_W1,   1536, 384, 0);
  pack(fw2,      OFF_W2,   384,  1536, 0);
  pack(gw,       OFF_WG,   384,  384, 0);

  const size_t shmem = 240 * 1024;   // 192KB multi-use + 48KB H1 (<= 320KB/WGP)
  (void)hipFuncSetAttribute((const void*)fused_attn_pool,
                            hipFuncAttributeMaxDynamicSharedMemorySize, (int)shmem);

  fused_attn_pool<<<32768 / 16, 256, shmem, stream>>>(
      x, latents, ca_b_in, ca_b_out, sa_b_in, sa_b_out,
      n1g, n1b, n2g, n2b, n3g, n3b, fb1, fb2, gb,
      wp, (float*)d_out);
}
